// NonLocalBlock2d_21861383537105
// MI455X (gfx1250) — matrix operand order repaired and checked by reading, not compiled, not run on hardware
//
#include <hip/hip_runtime.h>

// ---------------------------------------------------------------------------
// NonLocalBlock2d for MI455X (gfx1250): all contractions via
// v_wmma_f32_16x16x32_bf16 (wave32 WMMA), bf16 operands / fp32 accumulation.
// GEMM: 128x128 block tile, BK=64 stage, 8 waves, each wave 64x32 (4x2 WMMA
// tiles) -> 16 WMMAs per barrier round. A tile staged via async global->LDS
// (ASYNCcnt) when available; B tile software-pipelined through registers.
// ---------------------------------------------------------------------------

typedef __bf16 bf16;
typedef __bf16 v16bf __attribute__((ext_vector_type(16)));
typedef float  v8f   __attribute__((ext_vector_type(8)));
typedef int    v4i   __attribute__((vector_size(16)));

#define BM 128
#define BN 128
#define BK 64

#if defined(__has_builtin)
#if __has_builtin(__builtin_amdgcn_global_load_async_to_lds_b128)
#define USE_ASYNC_A 1
#endif
#endif
#ifndef USE_ASYNC_A
#define USE_ASYNC_A 0
#endif

__device__ __forceinline__ void async_wait_zero()
{
#if defined(__has_builtin)
#if __has_builtin(__builtin_amdgcn_s_wait_asynccnt)
    __builtin_amdgcn_s_wait_asynccnt(0);
    return;
#else
    asm volatile("s_wait_asynccnt 0" ::: "memory");
    return;
#endif
#else
    asm volatile("s_wait_asynccnt 0" ::: "memory");
#endif
}

__device__ __forceinline__ void load_b_tile(const bf16* __restrict__ bp, int ldb,
                                            uint4 br[4])
{
#pragma unroll
    for (int r = 0; r < 4; ++r)
        br[r] = *reinterpret_cast<const uint4*>(bp + (long)r * ldb);
}

// ---------------------------------------------------------------------------
// Generic batched GEMM: C[b] = alpha * A[b](MxK) * B[b](KxN) + bias[m]
// A,B bf16 row-major; C fp32 or bf16; optional transposed store.
// ---------------------------------------------------------------------------
__global__ __launch_bounds__(256)
void gemm_bf16_wmma(const bf16* __restrict__ A, long sA,
                    const bf16* __restrict__ B, long sB,
                    float* Cf, bf16* Cb, long sC,
                    int M, int N, int K,
                    int lda, int ldb, int ldc,
                    const float* __restrict__ bias,
                    float alpha, int transC)
{
    __shared__ bf16 lsA[BM][BK];    // A tile, row-major         (16 KB)
    __shared__ bf16 lsBt[BN][BK];   // B tile, transposed [n][k] (16 KB)

    const int tid  = threadIdx.x;
    const int lane = tid & 31;
    const int wid  = tid >> 5;
    const int wm   = (wid >> 2) * 64;   // wave row offset in block tile
    const int wn   = (wid & 3)  * 32;   // wave col offset in block tile

    const int  bn0 = blockIdx.x * BN;
    const int  bm0 = blockIdx.y * BM;
    const long bz  = blockIdx.z;

    const bf16* Ab = A + bz * sA;
    const bf16* Bb = B + bz * sB;

    const v8f vzero = {};
    v8f acc[4][2];
#pragma unroll
    for (int i = 0; i < 4; ++i)
#pragma unroll
        for (int j = 0; j < 2; ++j) acc[i][j] = vzero;

    // Cooperative-load coordinates
    const int arow = tid >> 1;          // 0..127 : A row in tile
    const int acb  = (tid & 1) * 32;    // 0/32   : A col half
    const int bkq  = tid >> 4;          // 0..15  : B k-quad (4 k rows)
    const int bcg  = (tid & 15) * 8;    // 0..120 : B col group (8 cols)

    // Fragment-load coordinates (ISA VGPR layouts, wave32)
    const int half = lane >> 4;         // 0 | 1
    const int lrow = lane & 15;
    const int akb  = half * 8;          // A: K chunks {0..7,16..23} / {8..15,24..31}
    const int bkb  = half * 8;          // B: K chunks {0..7,16..23} / {8..15,24..31}

    union Frag { struct { uint4 lo, hi; } q; v16bf v; };

    const bf16* aBase = Ab + (long)(bm0 + arow) * lda + acb;
    const bf16* bBase = Bb + (long)(4 * bkq) * ldb + bn0 + bcg;

    // ---- software pipeline prologue -------------------------------------
    uint4 brCur[4], brNxt[4];
    load_b_tile(bBase, ldb, brCur);
#if !USE_ASYNC_A
    uint4 arCur[4], arNxt[4];
#pragma unroll
    for (int c = 0; c < 4; ++c)
        arCur[c] = *reinterpret_cast<const uint4*>(aBase + 8 * c);
#endif

    for (int k0 = 0; k0 < K; k0 += BK) {
        // ---- stage A tile ------------------------------------------------
#if USE_ASYNC_A
        // async DMA: global -> LDS, 4 x 16B per thread, tracked by ASYNCcnt
#pragma unroll
        for (int c = 0; c < 4; ++c) {
            __builtin_amdgcn_global_load_async_to_lds_b128(
                (__attribute__((address_space(1))) v4i*)(aBase + k0 + 8 * c),
                (__attribute__((address_space(3))) v4i*)(&lsA[arow][acb + 8 * c]),
                0, 0);
        }
#else
#pragma unroll
        for (int c = 0; c < 4; ++c)
            *reinterpret_cast<uint4*>(&lsA[arow][acb + 8 * c]) = arCur[c];
#endif

        // ---- stage B tile transposed (pack 4 k-rows x 8 cols) ------------
        {
            const unsigned short* r0 = reinterpret_cast<const unsigned short*>(&brCur[0]);
            const unsigned short* r1 = reinterpret_cast<const unsigned short*>(&brCur[1]);
            const unsigned short* r2 = reinterpret_cast<const unsigned short*>(&brCur[2]);
            const unsigned short* r3 = reinterpret_cast<const unsigned short*>(&brCur[3]);
#pragma unroll
            for (int j = 0; j < 8; ++j) {
                uint2 w;
                w.x = (unsigned int)r0[j] | ((unsigned int)r1[j] << 16);
                w.y = (unsigned int)r2[j] | ((unsigned int)r3[j] << 16);
                *reinterpret_cast<uint2*>(&lsBt[bcg + j][4 * bkq]) = w;
            }
        }

        // ---- issue next-stage global loads (overlap with compute) --------
        const bool hasNext = (k0 + BK) < K;
        if (hasNext) {
            load_b_tile(bBase + (long)(k0 + BK) * ldb, ldb, brNxt);
#if !USE_ASYNC_A
#pragma unroll
            for (int c = 0; c < 4; ++c)
                arNxt[c] = *reinterpret_cast<const uint4*>(aBase + k0 + BK + 8 * c);
#endif
        }
        if (k0 + 2 * BK < K) {
            __builtin_prefetch(aBase + k0 + 2 * BK, 0, 1);
            __builtin_prefetch(bBase + (long)(k0 + 2 * BK) * ldb, 0, 1);
        }

#if USE_ASYNC_A
        async_wait_zero();
#endif
        __syncthreads();

        // ---- compute: 2 k-steps x (4 m-tiles x 2 n-tiles) = 16 WMMAs -----
#pragma unroll
        for (int ks = 0; ks < BK; ks += 32) {
            Frag fb[2];
#pragma unroll
            for (int j = 0; j < 2; ++j) {
                const bf16* p = &lsBt[wn + j * 16 + lrow][ks + bkb];
                fb[j].q.lo = *reinterpret_cast<const uint4*>(p);
                fb[j].q.hi = *reinterpret_cast<const uint4*>(p + 16);
            }
#pragma unroll
            for (int i = 0; i < 4; ++i) {
                Frag fa;
                const bf16* p = &lsA[wm + i * 16 + lrow][ks + akb];
                fa.q.lo = *reinterpret_cast<const uint4*>(p);
                fa.q.hi = *reinterpret_cast<const uint4*>(p + 16);
#pragma unroll
                for (int j = 0; j < 2; ++j)
                    acc[i][j] = __builtin_amdgcn_wmma_f32_16x16x32_bf16(
                        false, fa.v, false, fb[j].v,
                        (short)0, acc[i][j], false, false);
            }
        }
        __syncthreads();

        if (hasNext) {
#pragma unroll
            for (int r = 0; r < 4; ++r) brCur[r] = brNxt[r];
#if !USE_ASYNC_A
#pragma unroll
            for (int c = 0; c < 4; ++c) arCur[c] = arNxt[c];
#endif
        }
    }

    // ---- epilogue --------------------------------------------------------
    float* Cfb = Cf ? Cf + bz * sC : nullptr;
    bf16*  Cbb = Cb ? Cb + bz * sC : nullptr;
#pragma unroll
    for (int i = 0; i < 4; ++i) {
#pragma unroll
        for (int j = 0; j < 2; ++j) {
            const int mb = bm0 + wm + i * 16 + 8 * half;   // C layout: M = r + 8*half
            const int nb = bn0 + wn + j * 16 + lrow;       //           N = lane&15
#pragma unroll
            for (int r = 0; r < 8; ++r) {
                const int gm = mb + r;
                float v = acc[i][j][r] * alpha;
                if (bias) v += bias[gm];
                const long idx = transC ? (long)nb * ldc + gm
                                        : (long)gm * ldc + nb;
                if (Cfb) Cfb[idx] = v;
                else     Cbb[idx] = (bf16)v;
            }
        }
    }
}

// ---------------------------------------------------------------------------
// Elementwise / reduction helpers
// ---------------------------------------------------------------------------
__global__ void cvt_f32_bf16(const float* __restrict__ s, bf16* __restrict__ d, long n)
{
    long i = (long)blockIdx.x * blockDim.x + threadIdx.x;
    if (i < n) d[i] = (bf16)s[i];
}

__global__ void maxpool2x2_bf16(const float* __restrict__ x, bf16* __restrict__ xp)
{
    // x: [8,1024,64,64] -> xp: [8,1024,32,32] (bf16)
    long e = (long)blockIdx.x * blockDim.x + threadIdx.x;
    if (e >= (long)8 * 1024 * 32 * 32) return;
    int  wp = (int)(e & 31);
    int  hp = (int)((e >> 5) & 31);
    long bc = e >> 10;
    const float* p = x + (bc * 64 + 2 * hp) * 64 + 2 * wp;
    float m = fmaxf(fmaxf(p[0], p[1]), fmaxf(p[64], p[65]));
    xp[e] = (bf16)m;
}

__global__ __launch_bounds__(256)
void softmax_transpose(const bf16* __restrict__ S, bf16* __restrict__ attnT)
{
    // S: [B,4096,1024] (row n, softmax over 1024 cols m)
    // attnT: [B,1024,4096] -> attnT[b][m][n] = softmax(S[b][n][:])[m]
    const int n   = blockIdx.x;
    const int b   = blockIdx.y;
    const int tid = threadIdx.x;
    __shared__ float red[256];
    const bf16* row = S + ((long)b * 4096 + n) * 1024;

    float v[4];
    float mx = -3.0e38f;
#pragma unroll
    for (int i = 0; i < 4; ++i) {
        v[i] = (float)row[tid + 256 * i];
        mx = fmaxf(mx, v[i]);
    }
    red[tid] = mx; __syncthreads();
    for (int s = 128; s > 0; s >>= 1) {
        if (tid < s) red[tid] = fmaxf(red[tid], red[tid + s]);
        __syncthreads();
    }
    mx = red[0]; __syncthreads();

    float sum = 0.f;
#pragma unroll
    for (int i = 0; i < 4; ++i) { v[i] = __expf(v[i] - mx); sum += v[i]; }
    red[tid] = sum; __syncthreads();
    for (int s = 128; s > 0; s >>= 1) {
        if (tid < s) red[tid] += red[tid + s];
        __syncthreads();
    }
    const float inv = 1.f / red[0];

    bf16* out = attnT + (long)b * 1024 * 4096 + n;
#pragma unroll
    for (int i = 0; i < 4; ++i)
        out[(long)(tid + 256 * i) * 4096] = (bf16)(v[i] * inv);
}

__global__ __launch_bounds__(256)
void bn_stats(const float* __restrict__ z, float* __restrict__ mean, float* __restrict__ var)
{
    // z: [8,1024,4096]; one block per channel c, reduce over 8*4096 elems
    const int c   = blockIdx.x;
    const int tid = threadIdx.x;
    __shared__ float rs[256], rq[256];
    float s = 0.f, q = 0.f;
    for (int e = tid; e < 8 * 4096; e += 256) {
        int b = e >> 12, n = e & 4095;
        float v = z[((long)b * 1024 + c) * 4096 + n];
        s += v; q += v * v;
    }
    rs[tid] = s; rq[tid] = q; __syncthreads();
    for (int st = 128; st > 0; st >>= 1) {
        if (tid < st) { rs[tid] += rs[tid + st]; rq[tid] += rq[tid + st]; }
        __syncthreads();
    }
    if (tid == 0) {
        float m = rs[0] * (1.f / 32768.f);
        mean[c] = m;
        var[c]  = rq[0] * (1.f / 32768.f) - m * m;
    }
}

__global__ void bn_residual(const float* __restrict__ x, const float* __restrict__ z,
                            const float* __restrict__ mean, const float* __restrict__ var,
                            const float* __restrict__ gamma, const float* __restrict__ beta,
                            float* __restrict__ out)
{
    long i = (long)blockIdx.x * blockDim.x + threadIdx.x;
    if (i >= (long)8 * 1024 * 4096) return;
    int c = (int)((i >> 12) & 1023);
    float zn = (z[i] - mean[c]) * rsqrtf(var[c] + 1e-5f);
    out[i] = x[i] + zn * gamma[c] + beta[c];
}

// ---------------------------------------------------------------------------
// Host-side orchestration
// ---------------------------------------------------------------------------
extern "C" void kernel_launch(void* const* d_in, const int* in_sizes, int n_in,
                              void* d_out, int out_size, void* d_ws, size_t ws_size,
                              hipStream_t stream)
{
    (void)in_sizes; (void)n_in; (void)out_size; (void)ws_size;

    const float* x       = (const float*)d_in[0];
    const float* theta_w = (const float*)d_in[1];
    const float* theta_b = (const float*)d_in[2];
    const float* phi_w   = (const float*)d_in[3];
    const float* phi_b   = (const float*)d_in[4];
    const float* g_w     = (const float*)d_in[5];
    const float* g_b     = (const float*)d_in[6];
    const float* wz_w    = (const float*)d_in[7];
    const float* wz_b    = (const float*)d_in[8];
    const float* gamma   = (const float*)d_in[9];
    const float* beta    = (const float*)d_in[10];

    char* ws = (char*)d_ws;
    size_t off = 0;
    auto carve = [&](size_t bytes) -> char* {
        off = (off + 255) & ~(size_t)255;
        char* p = ws + off;
        off += bytes;
        return p;
    };

    const long NX = 8L * 1024 * 4096;        // 33.5M
    const long NP = 8L * 1024 * 1024;        // pooled

    bf16* xb     = (bf16*)carve(NX * 2);
    bf16* xpb    = (bf16*)carve(NP * 2);
    bf16* wthb   = (bf16*)carve(512L * 1024 * 2);
    bf16* wphb   = (bf16*)carve(512L * 1024 * 2);
    bf16* wgb    = (bf16*)carve(512L * 1024 * 2);
    bf16* wzb    = (bf16*)carve(1024L * 512 * 2);
    bf16* thetaT = (bf16*)carve(8L * 4096 * 512 * 2);
    bf16* phib   = (bf16*)carve(8L * 512 * 1024 * 2);
    bf16* gb     = (bf16*)carve(8L * 512 * 1024 * 2);
    bf16* Sb     = (bf16*)carve(8L * 4096 * 1024 * 2);
    bf16* attnT  = (bf16*)carve(8L * 1024 * 4096 * 2);
    bf16* yb     = (bf16*)carve(8L * 512 * 4096 * 2);
    float* zf    = (float*)carve(8L * 1024 * 4096 * 4);
    float* meanp = (float*)carve(1024 * 4);
    float* varp  = (float*)carve(1024 * 4);

    // 0) precision conversion + pooling
    cvt_f32_bf16<<<(unsigned)((NX + 255) / 256), 256, 0, stream>>>(x, xb, NX);
    cvt_f32_bf16<<<2048, 256, 0, stream>>>(theta_w, wthb, 512L * 1024);
    cvt_f32_bf16<<<2048, 256, 0, stream>>>(phi_w,   wphb, 512L * 1024);
    cvt_f32_bf16<<<2048, 256, 0, stream>>>(g_w,     wgb,  512L * 1024);
    cvt_f32_bf16<<<2048, 256, 0, stream>>>(wz_w,    wzb,  1024L * 512);
    maxpool2x2_bf16<<<(unsigned)((NP + 255) / 256), 256, 0, stream>>>(x, xpb);

    // 1) thetaT[b] = (theta_w @ x[b] + theta_b)^T   -> [4096,512]
    gemm_bf16_wmma<<<dim3(32, 4, 8), 256, 0, stream>>>(
        wthb, 0, xb, 1024L * 4096, nullptr, thetaT, 4096L * 512,
        512, 4096, 1024, 1024, 4096, 512, theta_b, 1.f, 1);

    // 2) phi[b] = phi_w @ xp[b] + phi_b   -> [512,1024]
    gemm_bf16_wmma<<<dim3(8, 4, 8), 256, 0, stream>>>(
        wphb, 0, xpb, 1024L * 1024, nullptr, phib, 512L * 1024,
        512, 1024, 1024, 1024, 1024, 1024, phi_b, 1.f, 0);

    // 3) g[b] = g_w @ xp[b] + g_b         -> [512,1024]
    gemm_bf16_wmma<<<dim3(8, 4, 8), 256, 0, stream>>>(
        wgb, 0, xpb, 1024L * 1024, nullptr, gb, 512L * 1024,
        512, 1024, 1024, 1024, 1024, 1024, g_b, 1.f, 0);

    // 4) S[b] = (thetaT[b] @ phi[b]) * C_INTER^-0.5  -> [4096,1024]
    gemm_bf16_wmma<<<dim3(8, 32, 8), 256, 0, stream>>>(
        thetaT, 4096L * 512, phib, 512L * 1024, nullptr, Sb, 4096L * 1024,
        4096, 1024, 512, 512, 1024, 1024, nullptr, 0.0441941738241592f, 0);

    // 5) row softmax over m + transpose -> attnT[b] : [1024,4096]
    softmax_transpose<<<dim3(4096, 8), 256, 0, stream>>>(Sb, attnT);

    // 6) y[b] = g[b] @ attnT[b]           -> [512,4096]
    gemm_bf16_wmma<<<dim3(32, 4, 8), 256, 0, stream>>>(
        gb, 512L * 1024, attnT, 1024L * 4096, nullptr, yb, 512L * 4096,
        512, 4096, 1024, 1024, 4096, 4096, nullptr, 1.f, 0);

    // 7) z[b] = wz_w @ y[b] + wz_b        -> [1024,4096] fp32
    gemm_bf16_wmma<<<dim3(32, 8, 8), 256, 0, stream>>>(
        wzb, 0, yb, 512L * 4096, zf, nullptr, 1024L * 4096,
        1024, 4096, 512, 512, 4096, 4096, wz_b, 1.f, 0);

    // 8) batchnorm stats + 9) normalize/affine/residual
    bn_stats<<<1024, 256, 0, stream>>>(zf, meanp, varp);
    bn_residual<<<(unsigned)((NX + 255) / 256), 256, 0, stream>>>(
        x, zf, meanp, varp, gamma, beta, (float*)d_out);
}
